// MambaPredictor_74543452389698
// MI455X (gfx1250) — matrix operand order repaired and checked by reading, not compiled, not run on hardware
//
#include <hip/hip_runtime.h>
#include <math.h>

// ---------------------------------------------------------------------------
// Types for CDNA5 WMMA (wave32): v_wmma_f32_16x16x32_bf16
// ---------------------------------------------------------------------------
typedef __attribute__((ext_vector_type(16))) __bf16 v16bf;
typedef __attribute__((ext_vector_type(8)))  float  v8f;

typedef union {
    v16bf        v;
    unsigned int u[8];
} FragBF;

#define BM 128
#define BN 64
#define BK 32
#define LROW 20   // u32 row stride in LDS: 16 bf16-pairs + 4 pad (80B, 16B-aligned)

#define CHUNK 128 // scan chunk length
#define NCH   16  // 2048 / CHUNK

// Single-instruction bf16 pair pack: dst = {bf16(y), bf16(x)} (hi, lo).
// Forced via inline asm so clang's bf16 vectorizer cannot re-associate the
// halves and emit mov_b16/and/or shuffle chains around the converts.
static __device__ __forceinline__ unsigned int pack_bf16(float x, float y) {
    unsigned int r;
    asm("v_cvt_pk_bf16_f32 %0, %1, %2" : "=v"(r) : "v"(x), "v"(y));
    return r;
}

struct StageRegs {
    float4 a[4];   // A tile: 128 rows x 16 pairs, 2 pairs (4 floats) per slot
    float4 b[2];   // B tile:  64 rows x 16 pairs
    bool aok[4];   // slot validity (dead code when KA && !NB)
    bool bok[2];
};

// ---------------------------------------------------------------------------
// Tiled GEMM: C[M,N] = A[M,K] (row-major, lda) x B[N,K]^T (row-major, ldb).
// f32 in memory; converted to packed bf16 during LDS staging; f32 accumulate.
// Block tile 128x64, 256 threads = 8 waves, each wave computes 32x32
// (2x2 fragments -> 4 WMMAs per K-step with A/B fragment reuse).
// Two-stage software pipeline with double-buffered LDS:
//   load(k+1) -> compute(k) -> store(k+1) -> barrier
// plus depth-2 speculative prefetch (global_prefetch_b8) of slab k+2.
// EPI: 0 = store, 1 = +bias[n], 2 = softplus(acc+bias[n]), 3 = C += acc
// NB:  true if N is not a multiple of BN (adds N-bounds guards).
// KA:  true if K is a multiple of BK (drops all K-guards; 4 of 5 GEMMs).
// Requires M % 128 == 0 (true here: M = 16384) and lda/ldb % 4 == 0.
// ---------------------------------------------------------------------------
template <int EPI, bool NB, bool KA>
__global__ __launch_bounds__(256)
void wmma_gemm_kernel(const float* __restrict__ A, const float* __restrict__ Bw,
                      float* __restrict__ C, const float* __restrict__ bias,
                      int M, int N, int K, int lda, int ldb, int ldc)
{
    __shared__ unsigned int As_u[2 * BM * LROW];   // 2 x 10240 B
    __shared__ unsigned int Bs_u[2 * BN * LROW];   // 2 x  5120 B

    const int tid   = threadIdx.x;
    const int lane  = tid & 31;
    const int wave  = tid >> 5;
    const int wm    = wave & 3;      // 0..3 -> rows 32*wm
    const int wn    = wave >> 2;     // 0..1 -> cols 32*wn
    const int lmod  = lane & 15;
    const int lhalf = lane >> 4;

    const int m_base = blockIdx.y * BM;
    const int n_base = blockIdx.x * BN;

    v8f c00 = {0.f,0.f,0.f,0.f,0.f,0.f,0.f,0.f};
    v8f c01 = c00, c10 = c00, c11 = c00;

    // Phase 1: issue ALL global loads for one K-slab into registers (batched,
    // b128 loads, no intervening waits -> full memory-level parallelism),
    // and speculatively prefetch the slab after it (same rows, +BK floats).
    auto stage_load = [&](int k0, StageRegs& sr) {
        #pragma unroll
        for (int i = 0; i < 4; ++i) {            // A: 2048 pairs / 2 per slot
            int p  = 2 * tid + i * 512;          // even pair index
            int r  = p >> 4;                     // row 0..127
            int pc = p & 15;                     // even pair column
            int gk = k0 + 2 * pc;                // K%4==0 -> quad all-or-nothing
            bool ok = KA ? true : (gk < K);
            sr.aok[i] = ok;
            size_t off = ok ? ((size_t)(m_base + r) * lda + gk) : 0;
            sr.a[i] = *(const float4*)(A + off);
            __builtin_prefetch(A + off + BK, 0, 1);   // depth-2, speculative
        }
        #pragma unroll
        for (int i = 0; i < 2; ++i) {            // B: 1024 pairs / 2 per slot
            int p  = 2 * tid + i * 512;
            int r  = p >> 4;                     // row 0..63
            int pc = p & 15;
            int gn = n_base + r;
            int gk = k0 + 2 * pc;
            bool ok = (!NB || gn < N) && (KA || gk < K);
            sr.bok[i] = ok;
            size_t off = ok ? ((size_t)gn * ldb + gk) : 0;
            sr.b[i] = *(const float4*)(Bw + off);
            __builtin_prefetch(Bw + off + BK, 0, 1);  // depth-2, speculative
        }
    };

    // Phase 2: convert to packed bf16 (one v_cvt_pk_bf16_f32 per pair),
    // mask AFTER packing (2 u32 selects, only when guards can fire),
    // store wide (b64) into the LDS buffer.
    auto stage_store = [&](const StageRegs& sr, int sel) {
        unsigned int* Ad = As_u + sel * (BM * LROW);
        unsigned int* Bd = Bs_u + sel * (BN * LROW);
        #pragma unroll
        for (int i = 0; i < 4; ++i) {
            int p  = 2 * tid + i * 512;
            int r  = p >> 4;
            int pc = p & 15;
            uint2 w;
            w.x = pack_bf16(sr.a[i].x, sr.a[i].y);
            w.y = pack_bf16(sr.a[i].z, sr.a[i].w);
            if (!KA) {
                w.x = sr.aok[i] ? w.x : 0u;
                w.y = sr.aok[i] ? w.y : 0u;
            }
            *(uint2*)(Ad + r * LROW + pc) = w;   // 8B-aligned -> ds_store_b64
        }
        #pragma unroll
        for (int i = 0; i < 2; ++i) {
            int p  = 2 * tid + i * 512;
            int r  = p >> 4;
            int pc = p & 15;
            uint2 w;
            w.x = pack_bf16(sr.b[i].x, sr.b[i].y);
            w.y = pack_bf16(sr.b[i].z, sr.b[i].w);
            if (!KA || NB) {
                w.x = sr.bok[i] ? w.x : 0u;
                w.y = sr.bok[i] ? w.y : 0u;
            }
            *(uint2*)(Bd + r * LROW + pc) = w;
        }
    };

    StageRegs sr;
    stage_load(0, sr);
    stage_store(sr, 0);
    __syncthreads();

    int cur = 0;
    for (int k0 = 0; k0 < K; k0 += BK) {
        const bool more = (k0 + BK) < K;
        StageRegs nx;
        if (more) stage_load(k0 + BK, nx);      // global loads in flight...

        const unsigned int* Au = As_u + cur * (BM * LROW);
        const unsigned int* Bu = Bs_u + cur * (BN * LROW);

        // fragments in contiguous k order (k = 16*lhalf + s), same for A and B:
        // A 16x32 bf16: lane(0-15)=row, lanes 16-31 same rows K+16.
        // B 32x16 bf16: lane(0-15)=col, lanes 16-31 same cols K+16.
        FragBF a0, a1, b0, b1;
        const int ar0 = (32 * wm + lmod) * LROW;
        const int ar1 = ar0 + 16 * LROW;
        const int br0 = (32 * wn + lmod) * LROW;
        const int br1 = br0 + 16 * LROW;
        #pragma unroll
        for (int v = 0; v < 8; ++v) {
            const int ia = 8 * lhalf + v;
            const int ib = 8 * lhalf + v;
            a0.u[v] = Au[ar0 + ia];
            a1.u[v] = Au[ar1 + ia];
            b0.u[v] = Bu[br0 + ib];
            b1.u[v] = Bu[br1 + ib];
        }

        c00 = __builtin_amdgcn_wmma_f32_16x16x32_bf16(false, a0.v, false, b0.v,
                                                      (short)0, c00, false, false);
        c01 = __builtin_amdgcn_wmma_f32_16x16x32_bf16(false, a0.v, false, b1.v,
                                                      (short)0, c01, false, false);
        c10 = __builtin_amdgcn_wmma_f32_16x16x32_bf16(false, a1.v, false, b0.v,
                                                      (short)0, c10, false, false);
        c11 = __builtin_amdgcn_wmma_f32_16x16x32_bf16(false, a1.v, false, b1.v,
                                                      (short)0, c11, false, false);

        if (more) stage_store(nx, cur ^ 1);     // ...landed behind the WMMAs
        __syncthreads();
        cur ^= 1;
    }

    // ---- epilogue. D layout: VGPR r -> M = r + 8*lhalf; lane(0-15) -> N ----
    const int gn0 = n_base + 32 * wn + lmod;
    const int gn1 = gn0 + 16;
    float bv0 = 0.f, bv1 = 0.f;
    if (EPI == 1 || EPI == 2) {
        bv0 = bias[(NB && gn0 >= N) ? 0 : gn0];
        bv1 = bias[(NB && gn1 >= N) ? 0 : gn1];
    }
    auto emit = [&](int gm, int gn, float v, float bv) {
        if (NB && gn >= N) return;
        size_t off = (size_t)gm * ldc + gn;
        if (EPI == 1) {
            v += bv;
        } else if (EPI == 2) {
            v += bv;
            v = (v > 20.f) ? v : log1pf(__expf(v));   // softplus
        } else if (EPI == 3) {
            v += C[off];                               // residual accumulate
        }
        C[off] = v;
    };
    #pragma unroll
    for (int r = 0; r < 8; ++r) {
        const int gm0 = m_base + 32 * wm + r + 8 * lhalf;
        const int gm1 = gm0 + 16;
        emit(gm0, gn0, c00[r], bv0);
        emit(gm0, gn1, c01[r], bv1);
        emit(gm1, gn0, c10[r], bv0);
        emit(gm1, gn1, c11[r], bv1);
    }
}

// ---------------------------------------------------------------------------
// RMSNorm: one block per row of 768
// ---------------------------------------------------------------------------
__global__ __launch_bounds__(256)
void rmsnorm_kernel(const float* __restrict__ x, const float* __restrict__ w,
                    float* __restrict__ out)
{
    __shared__ float red[256];
    const int row = blockIdx.x;
    const int tid = threadIdx.x;
    const float* xr = x + (size_t)row * 768;
    float ss = 0.f;
    for (int c = tid; c < 768; c += 256) { float v = xr[c]; ss += v * v; }
    red[tid] = ss; __syncthreads();
    for (int s = 128; s > 0; s >>= 1) {
        if (tid < s) red[tid] += red[tid + s];
        __syncthreads();
    }
    const float scale = rsqrtf(red[0] * (1.f / 768.f) + 1e-5f);
    float* orow = out + (size_t)row * 768;
    for (int c = tid; c < 768; c += 256) orow[c] = xr[c] * scale * w[c];
}

// ---------------------------------------------------------------------------
// Causal depthwise conv1d (width 4, left-pad 3) + bias + SiLU.
// ---------------------------------------------------------------------------
__global__ __launch_bounds__(256)
void conv_silu_kernel(const float* __restrict__ xz, const float* __restrict__ cw,
                      const float* __restrict__ cb, float* __restrict__ u)
{
    const size_t idx = (size_t)blockIdx.x * 256 + threadIdx.x;  // b*L*1536
    const int d = (int)(idx % 1536);
    const size_t bt = idx / 1536;           // b*2048 + t
    const int t = (int)(bt % 2048);
    float acc = cb[d];
    #pragma unroll
    for (int j = 0; j < 4; ++j) {
        int tt = t - 3 + j;
        size_t src = (bt - 3 + j) * 3072 + d;
        float xv = xz[tt >= 0 ? src : (size_t)d];   // always-load, masked
        acc += (tt >= 0 ? cw[d * 4 + j] : 0.f) * xv;
    }
    u[idx] = acc / (1.f + __expf(-acc));    // silu
}

// ---------------------------------------------------------------------------
// Selective scan, chunked-parallel over L (associative recurrence
// h_t = a_t*h_{t-1} + b_t). Pass 1: per-chunk scan from zero state, record
// chunk product of a's and chunk-final partial state. Pass 2: tiny serial
// combine over NCH chunks -> per-chunk initial states (written into pA).
// Pass 3: re-run each chunk from its true initial state, emit gated output.
// ---------------------------------------------------------------------------
__global__ __launch_bounds__(256)
void scan_pass1(const float* __restrict__ delta, const float* __restrict__ u,
                const float* __restrict__ dbc, const float* __restrict__ A_log,
                float* __restrict__ hC, float* __restrict__ pA)
{
    const int gid = blockIdx.x * 256 + threadIdx.x;   // 0 .. 12288*NCH-1
    const int d   = gid % 1536;
    const int rem = gid / 1536;
    const int b   = rem & 7;
    const int c   = rem >> 3;

    float Ac[16];
    #pragma unroll
    for (int n = 0; n < 16; ++n) Ac[n] = -__expf(A_log[d * 16 + n]);
    float h[16], p[16];
    #pragma unroll
    for (int n = 0; n < 16; ++n) { h[n] = 0.f; p[n] = 1.f; }

    const int t0 = c * CHUNK;
    for (int t = t0; t < t0 + CHUNK; ++t) {
        const size_t row = (size_t)b * 2048 + t;
        const float dl = delta[row * 1536 + d];
        const float ut = u[row * 1536 + d];
        const float* bc = dbc + row * 80;
        const float du = dl * ut;
        #pragma unroll
        for (int n = 0; n < 16; ++n) {
            float a = __expf(dl * Ac[n]);
            h[n] = a * h[n] + du * bc[48 + n];
            p[n] *= a;
        }
    }
    const size_t base = ((size_t)(c * 8 + b) * 1536 + d) * 16;
    #pragma unroll
    for (int n = 0; n < 16; ++n) { hC[base + n] = h[n]; pA[base + n] = p[n]; }
}

__global__ __launch_bounds__(256)
void scan_pass2(const float* __restrict__ hC, float* __restrict__ pA)
{
    const int gid = blockIdx.x * 256 + threadIdx.x;   // 0 .. 196607
    const int n = gid & 15;
    const int d = (gid >> 4) % 1536;
    const int b = gid / (16 * 1536);
    float H = 0.f;
    for (int c = 0; c < NCH; ++c) {
        const size_t idx = ((size_t)(c * 8 + b) * 1536 + d) * 16 + n;
        const float hc = hC[idx];
        const float pa = pA[idx];
        pA[idx] = H;                 // per-chunk initial state
        H = pa * H + hc;
    }
}

__global__ __launch_bounds__(256)
void scan_pass3(const float* __restrict__ delta, const float* __restrict__ xz,
                const float* __restrict__ dbc, const float* __restrict__ A_log,
                const float* __restrict__ Dp, const float* __restrict__ Hinit,
                float* __restrict__ u_io)
{
    const int gid = blockIdx.x * 256 + threadIdx.x;
    const int d   = gid % 1536;
    const int rem = gid / 1536;
    const int b   = rem & 7;
    const int c   = rem >> 3;

    float Ac[16];
    #pragma unroll
    for (int n = 0; n < 16; ++n) Ac[n] = -__expf(A_log[d * 16 + n]);
    const float Dval = Dp[d];

    float h[16];
    const size_t base = ((size_t)(c * 8 + b) * 1536 + d) * 16;
    #pragma unroll
    for (int n = 0; n < 16; ++n) h[n] = Hinit[base + n];

    const int t0 = c * CHUNK;
    for (int t = t0; t < t0 + CHUNK; ++t) {
        const size_t row = (size_t)b * 2048 + t;
        const float dl = delta[row * 1536 + d];
        const float ut = u_io[row * 1536 + d];
        const float* bc = dbc + row * 80;
        const float du = dl * ut;
        float y = 0.f;
        #pragma unroll
        for (int n = 0; n < 16; ++n) {
            h[n] = __expf(dl * Ac[n]) * h[n] + du * bc[48 + n];
            y += h[n] * bc[64 + n];
        }
        y += ut * Dval;
        const float r = xz[row * 3072 + 1536 + d];      // res gate
        y *= r / (1.f + __expf(-r));                    // y * silu(res)
        u_io[row * 1536 + d] = y;
    }
}

// ---------------------------------------------------------------------------
// Final RMSNorm on last token + output projection (OUT_DIM=1).
// ---------------------------------------------------------------------------
__global__ __launch_bounds__(256)
void head_kernel(const float* __restrict__ X, const float* __restrict__ nw,
                 const float* __restrict__ W_out, const float* __restrict__ b_out,
                 float* __restrict__ out)
{
    __shared__ float red[256];
    const int b = blockIdx.x;
    const int tid = threadIdx.x;
    const float* row = X + ((size_t)b * 2048 + 2047) * 768;
    float ss = 0.f;
    for (int c = tid; c < 768; c += 256) { float v = row[c]; ss += v * v; }
    red[tid] = ss; __syncthreads();
    for (int s = 128; s > 0; s >>= 1) {
        if (tid < s) red[tid] += red[tid + s];
        __syncthreads();
    }
    const float scale = rsqrtf(red[0] * (1.f / 768.f) + 1e-5f);
    __syncthreads();
    float dot = 0.f;
    for (int c = tid; c < 768; c += 256) dot += row[c] * scale * nw[c] * W_out[c];
    red[tid] = dot; __syncthreads();
    for (int s = 128; s > 0; s >>= 1) {
        if (tid < s) red[tid] += red[tid + s];
        __syncthreads();
    }
    if (tid == 0) out[b] = red[0] + b_out[0];
}

// ---------------------------------------------------------------------------
// Launcher
// ---------------------------------------------------------------------------
extern "C" void kernel_launch(void* const* d_in, const int* in_sizes, int n_in,
                              void* d_out, int out_size, void* d_ws, size_t ws_size,
                              hipStream_t stream)
{
    (void)in_sizes; (void)n_in; (void)out_size; (void)ws_size;

    const float* features   = (const float*)d_in[0];
    const float* W_in       = (const float*)d_in[1];
    const float* b_in       = (const float*)d_in[2];
    const float* in_proj_w  = (const float*)d_in[3];
    const float* conv_w     = (const float*)d_in[4];
    const float* conv_b     = (const float*)d_in[5];
    const float* x_proj_w   = (const float*)d_in[6];
    const float* dt_proj_w  = (const float*)d_in[7];
    const float* dt_proj_b  = (const float*)d_in[8];
    const float* A_log      = (const float*)d_in[9];
    const float* D_param    = (const float*)d_in[10];
    const float* out_proj_w = (const float*)d_in[11];
    const float* norm_w     = (const float*)d_in[12];
    const float* norm_f_w   = (const float*)d_in[13];
    const float* W_out      = (const float*)d_in[14];
    const float* b_out      = (const float*)d_in[15];

    const int M = 8 * 2048;   // 16384 rows (B*L); multiple of BM=128

    float* X     = (float*)d_ws;                  // (M, 768)  residual stream
    float* XN    = X     + (size_t)M * 768;       // (M, 768)  normed input
    float* XZ    = XN    + (size_t)M * 768;       // (M, 3072) in_proj out
    float* U     = XZ    + (size_t)M * 3072;      // (M, 1536) conv out / gated y
    float* DBC   = U     + (size_t)M * 1536;      // (M, 80)   x_proj out
    float* DELTA = DBC   + (size_t)M * 80;        // (M, 1536) softplus(dt)
    float* HC    = DELTA + (size_t)M * 1536;      // (NCH*8, 1536, 16) chunk states
    float* PA    = HC    + (size_t)NCH * 8 * 1536 * 16; // chunk a-prod / Hinit

    dim3 blk(256);
    const int scan_blocks = (12288 * NCH) / 256;   // 768
    const int comb_blocks = (8 * 1536 * 16) / 256; // 768

    auto grid_for = [&](int N) { return dim3((N + BN - 1) / BN, M / BM); };

    // input embedding: X = features @ W_in^T + b_in
    wmma_gemm_kernel<1, false, true><<<grid_for(768), blk, 0, stream>>>(
        features, W_in, X, b_in, M, 768, 64, 64, 64, 768);

    for (int l = 0; l < 4; ++l) {
        // XN = rmsnorm(X, norm_w[l])
        rmsnorm_kernel<<<M, blk, 0, stream>>>(X, norm_w + (size_t)l * 768, XN);
        // XZ = XN @ in_proj_w[l]^T
        wmma_gemm_kernel<0, false, true><<<grid_for(3072), blk, 0, stream>>>(
            XN, in_proj_w + (size_t)l * 3072 * 768, XZ, nullptr,
            M, 3072, 768, 768, 768, 3072);
        // U = silu(causal_conv(XZ[:, :1536]) + conv_b)
        conv_silu_kernel<<<(M * 1536) / 256, blk, 0, stream>>>(
            XZ, conv_w + (size_t)l * 1536 * 4, conv_b + (size_t)l * 1536, U);
        // DBC = U @ x_proj_w[l]^T   (dt | B | C)
        wmma_gemm_kernel<0, true, true><<<grid_for(80), blk, 0, stream>>>(
            U, x_proj_w + (size_t)l * 80 * 1536, DBC, nullptr,
            M, 80, 1536, 1536, 1536, 80);
        // DELTA = softplus(DBC[:, :48] @ dt_proj_w[l]^T + dt_proj_b[l])
        wmma_gemm_kernel<2, false, false><<<grid_for(1536), blk, 0, stream>>>(
            DBC, dt_proj_w + (size_t)l * 1536 * 48, DELTA,
            dt_proj_b + (size_t)l * 1536, M, 1536, 48, 80, 48, 1536);
        // chunked selective scan + D skip + silu gate, in place into U
        scan_pass1<<<scan_blocks, blk, 0, stream>>>(
            DELTA, U, DBC, A_log + (size_t)l * 1536 * 16, HC, PA);
        scan_pass2<<<comb_blocks, blk, 0, stream>>>(HC, PA);
        scan_pass3<<<scan_blocks, blk, 0, stream>>>(
            DELTA, XZ, DBC, A_log + (size_t)l * 1536 * 16,
            D_param + (size_t)l * 1536, PA, U);
        // X += U @ out_proj_w[l]^T
        wmma_gemm_kernel<3, false, true><<<grid_for(768), blk, 0, stream>>>(
            U, out_proj_w + (size_t)l * 768 * 1536, X, nullptr,
            M, 768, 1536, 1536, 1536, 768);
    }

    // final rmsnorm (last token) + head
    head_kernel<<<8, blk, 0, stream>>>(X, norm_f_w, W_out, b_out, (float*)d_out);
}
